// PhyChemPPI_34531537060474
// MI455X (gfx1250) — matrix operand order repaired and checked by reading, not compiled, not run on hardware
//
#include <hip/hip_runtime.h>

typedef __attribute__((ext_vector_type(16))) _Float16 v16h;
typedef __attribute__((ext_vector_type(8)))  float    v8f;

#define S_SEQ 430
#define BT    32
#define KS1   11   // fc1 K padded to 352 = 11*32
#define KS2   27   // head K padded to 864 = 27*32

#define GLOBAL_AS __attribute__((address_space(1)))
#define LDS_AS    __attribute__((address_space(3)))

// ---- WMMA LDS swizzle helpers (ISA 7.12.2, 16-bit, wave32) ----
// A 16x32 f16: lane<16 row m=lane: i0..7->K=i, i8..15->K=i+8; lanes16-31: +8 / +16
// layout: [mtile][kstep][lane 0..31][16 halfs] -> 32B contiguous per lane
__device__ __forceinline__ int a_swz(int m, int k, int ksteps) {
    int mt = m >> 4, ml = m & 15;
    int kk = k >> 5, r = k & 31;
    int lane = ml + (((r >> 3) & 1) << 4);
    int i    = (r & 7) + (((r >> 4) & 1) << 3);
    return ((mt * ksteps + kk) * 32 + lane) * 16 + i;
}
// B 32x16 f16: lane<16 col n=lane: i0..7->K=i, i8..15->K=i+8; lanes16-31: +8 / +16
__device__ __forceinline__ int b_swz(int n, int k, int ksteps) {
    int nt = n >> 4, nl = n & 15;
    int kk = k >> 5, r = k & 31;
    int lane = nl + (((r >> 3) & 1) << 4);
    int i    = (r & 7) + (((r >> 4) & 1) << 3);
    return ((nt * ksteps + kk) * 32 + lane) * 16 + i;
}

__device__ __forceinline__ v8f wmma16(const _Float16* Ap, const _Float16* Bp, v8f acc) {
    v16h a = *(const v16h*)Ap;
    v16h b = *(const v16h*)Bp;
    return __builtin_amdgcn_wmma_f32_16x16x32_f16(false, a, false, b, (short)0, acc,
                                                  false, false);
}

// ================= Kernel 1: per-sequence conv+fc stack, all-WMMA =================
__global__ __launch_bounds__(256, 2)
void seq_net_kernel(const float* __restrict__ X,
                    const float* __restrict__ cw1, const float* __restrict__ cb1,
                    const float* __restrict__ cw2, const float* __restrict__ cb2,
                    const float* __restrict__ fw1, const float* __restrict__ fb1,
                    const float* __restrict__ fw2, const float* __restrict__ fb2,
                    const float* __restrict__ fw3, const float* __restrict__ fb3,
                    float* __restrict__ gcat)
{
    // Phase-scoped LDS regions (reuse keeps total ~118 KB -> 2 blocks/WGP)
    __shared__ __align__(32) char RA[49152]; // A1 (768x32 im2col) | {c2out 22528 + fc1 Abuf 22528}
    __shared__ __align__(32) char RB[48128]; // Xs 32x376 f32      | {A2 (704x32) | Bw1 (64x352)}
    __shared__ __align__(32) _Float16 h1f[BT][8][24];       // conv1 relu out (12 KB)
    __shared__ __align__(32) _Float16 Bc1[512];             // conv1 W  [32k x 16n]
    __shared__ __align__(32) _Float16 Bc2[512];             // conv2 W  [32k x 16n]
    __shared__ __align__(32) _Float16 A2fc[2 * 2 * 32 * 16];// fc1 out [32 x 64]
    __shared__ __align__(32) _Float16 B2[2 * 32 * 16];      // fw2 [16 x 64]
    __shared__ float f3[BT][16];
    __shared__ float cb1s[8], cb2s[16], fb1s[64], fb2s[10], fw3s[2][10], fb3s[2];

    _Float16* A1   = (_Float16*)RA;            // phase 1-2
    _Float16* c2p  = (_Float16*)RA;            // phase 4-4b: [e][w2(22)][o(16)]
    _Float16* Abuf = (_Float16*)(RA + 22528);  // phase 4b-5: fc1 A [2mt][11][32][16]
    float*    Xs   = (float*)RB;               // phase 0-1: [32][376]
    _Float16* A2   = (_Float16*)RB;            // phase 3-4
    _Float16* Bw1  = (_Float16*)RB;            // phase 4b-5: fw1 B [4nt][11][32][16]

    const int tid  = threadIdx.x;
    const int lane = tid & 31;
    const int wave = tid >> 5;
    const int s    = blockIdx.x;
    const int b0   = blockIdx.y * BT;

    // ---------- phase 0: weights + X tile ----------
    if (tid < 8)  cb1s[tid] = cb1[s * 8 + tid];
    if (tid < 16) cb2s[tid] = cb2[s * 16 + tid];
    if (tid < 64) fb1s[tid] = fb1[s * 64 + tid];
    if (tid < 10) fb2s[tid] = fb2[s * 10 + tid];
    if (tid < 20) fw3s[tid / 10][tid % 10] = fw3[s * 20 + tid];
    if (tid < 2)  fb3s[tid] = fb3[s * 2 + tid];

    // conv1 weights -> B layout: k = h*2+d (30 valid), n = c (8 valid)
    for (int idx = tid; idx < 512; idx += 256) {
        int n = idx >> 5, k = idx & 31;
        float v = (n < 8 && k < 30) ? cw1[((s * 8 + n) * 15 + (k >> 1)) * 2 + (k & 1)] : 0.f;
        Bc1[b_swz(n, k, 1)] = (_Float16)v;
    }
    // conv2 weights -> B layout: k = c*2+d (16 valid), n = o (16 valid)
    for (int idx = tid; idx < 512; idx += 256) {
        int n = idx >> 5, k = idx & 31;
        float v = (k < 16) ? cw2[((s * 16 + n) * 8 + (k >> 1)) * 2 + (k & 1)] : 0.f;
        Bc2[b_swz(n, k, 1)] = (_Float16)v;
    }
    // fw2 -> B layout (16 x 64, rows >= 10 zero)
    for (int idx = tid; idx < 1024; idx += 256) {
        int n = idx >> 6, k = idx & 63;
        float v = (n < 10) ? fw2[(size_t)(s * 10 + n) * 64 + k] : 0.f;
        B2[b_swz(n, k, 2)] = (_Float16)v;
    }
    // stage X tile (coalesced, one wave per row). X is a 330 MB one-pass stream:
    // async Global->LDS (ASYNCcnt path) if available, else non-temporal loads.
#if __has_builtin(__builtin_amdgcn_global_load_async_to_lds_b32)
    for (int e = wave; e < BT; e += 8) {
        const float* src = X + ((size_t)(b0 + e) * S_SEQ + s) * 375;
        for (int j = lane; j < 375; j += 32)
            __builtin_amdgcn_global_load_async_to_lds_b32(
                (GLOBAL_AS int*)(src + j),
                (LDS_AS int*)(Xs + e * 376 + j), 0, 0);
    }
  #if __has_builtin(__builtin_amdgcn_s_wait_asynccnt)
    __builtin_amdgcn_s_wait_asynccnt(0);
  #else
    asm volatile("s_wait_asynccnt 0x0" ::: "memory");
  #endif
#else
    for (int e = wave; e < BT; e += 8) {
        const float* src = X + ((size_t)(b0 + e) * S_SEQ + s) * 375;
        for (int j = lane; j < 375; j += 32)
            Xs[e * 376 + j] = __builtin_nontemporal_load(src + j);
    }
#endif
    __syncthreads();

    // ---------- phase 1: im2col for conv1: A1[m'=e*24+w][k=h*2+d] ----------
    #pragma unroll
    for (int rr = 0; rr < 3; ++rr) {
        int mp = tid * 3 + rr;                 // 768 rows
        int e = mp / 24, w = mp - e * 24;
        #pragma unroll
        for (int k = 0; k < 32; ++k) {
            float v = (k < 30) ? Xs[e * 376 + (k >> 1) * 25 + w + (k & 1)] : 0.f;
            A1[a_swz(mp, k, 1)] = (_Float16)v;
        }
    }
    __syncthreads();

    // ---------- phase 2: conv1 WMMA (48 tiles) + bias/relu -> h1f ----------
    for (int i = 0; i < 6; ++i) {
        int mt = wave + 8 * i;
        v8f acc = {0.f, 0.f, 0.f, 0.f, 0.f, 0.f, 0.f, 0.f};
        acc = wmma16(&A1[(mt * 32 + lane) * 16], &Bc1[lane * 16], acc);
        int c = lane & 15, mh = (lane >> 4) << 3;
        if (c < 8) {
            #pragma unroll
            for (int r = 0; r < 8; ++r) {
                int mp = mt * 16 + mh + r;
                int e = mp / 24, w = mp - e * 24;
                h1f[e][c][w] = (_Float16)fmaxf(acc[r] + cb1s[c], 0.f);
            }
        }
    }
    __syncthreads();

    // ---------- phase 3: pool1 + im2col for conv2: A2[m''=e*22+w2][k=c*2+d] ----------
    #pragma unroll
    for (int rr = 0; rr < 3; ++rr) {
        int mp = tid * 3 + rr;                 // 704 rows
        if (mp < 704) {
            int e = mp / 22, w2 = mp - e * 22;
            #pragma unroll
            for (int k = 0; k < 32; ++k) {
                float v = 0.f;
                if (k < 16) {
                    int c = k >> 1, d = k & 1;
                    v = fmaxf((float)h1f[e][c][w2 + d], (float)h1f[e][c][w2 + d + 1]);
                }
                A2[a_swz(mp, k, 1)] = (_Float16)v;
            }
        }
    }
    __syncthreads();

    // ---------- phase 4: conv2 WMMA (44 tiles) + bias/relu -> c2p ----------
    for (int i = 0; i < 6; ++i) {
        int mt = wave + 8 * i;                 // wave-uniform guard: EXEC stays full
        if (mt < 44) {
            v8f acc = {0.f, 0.f, 0.f, 0.f, 0.f, 0.f, 0.f, 0.f};
            acc = wmma16(&A2[(mt * 32 + lane) * 16], &Bc2[lane * 16], acc);
            int o = lane & 15, mh = (lane >> 4) << 3;
            #pragma unroll
            for (int r = 0; r < 8; ++r) {
                int mp = mt * 16 + mh + r;
                int e = mp / 22, w2 = mp - e * 22;
                c2p[(e * 22 + w2) * 16 + o] = (_Float16)fmaxf(acc[r] + cb2s[o], 0.f);
            }
        }
    }
    __syncthreads();

    // ---------- phase 4b: pool2+flatten -> Abuf ; load fw1 -> Bw1 ----------
    {   // Abuf: m = e, k = o*21 + w2 (channel-major flatten), pad K to 352
        int e = tid >> 3, j = tid & 7;
        #pragma unroll
        for (int oo = 0; oo < 2; ++oo) {
            int o = 2 * j + oo;
            for (int w2 = 0; w2 < 21; ++w2) {
                float v = fmaxf((float)c2p[(e * 22 + w2) * 16 + o],
                                (float)c2p[(e * 22 + w2 + 1) * 16 + o]);
                Abuf[a_swz(e, o * 21 + w2, KS1)] = (_Float16)v;
            }
        }
        if (j == 0)
            for (int k = 336; k < 352; ++k) Abuf[a_swz(e, k, KS1)] = (_Float16)0.f;
    }
    {   // Bw1: 64 x 352 (K zero-padded); hot in L2 across the 16 batch tiles
        int n = tid >> 2, j = tid & 3;
        const float* wrow = fw1 + (size_t)(s * 64 + n) * 336;
        for (int k = j * 88; k < j * 88 + 88; ++k) {
            float v = (k < 336) ? wrow[k] : 0.f;
            Bw1[b_swz(n, k, KS1)] = (_Float16)v;
        }
    }
    __syncthreads();

    // ---------- phase 5: fc1 WMMA [32x352]x[352x64], 1 tile/wave ----------
    {
        int mt = wave >> 2, nt = wave & 3;
        v8f acc = {0.f, 0.f, 0.f, 0.f, 0.f, 0.f, 0.f, 0.f};
        #pragma unroll
        for (int kk = 0; kk < KS1; ++kk)
            acc = wmma16(&Abuf[((mt * KS1 + kk) * 32 + lane) * 16],
                         &Bw1 [((nt * KS1 + kk) * 32 + lane) * 16], acc);
        int nl = lane & 15, mh = (lane >> 4) << 3;
        int n = nt * 16 + nl;
        #pragma unroll
        for (int r = 0; r < 8; ++r) {
            int m = mt * 16 + mh + r;
            float v = fmaxf(acc[r] + fb1s[n], 0.f);
            A2fc[a_swz(m, n, 2)] = (_Float16)v;   // fc2 input, K index = n
        }
    }
    __syncthreads();

    // ---------- phase 6: fc2 WMMA [32x64]x[64x16(pad of 10)] ----------
    if (wave < 2) {
        int mt = wave;
        v8f acc = {0.f, 0.f, 0.f, 0.f, 0.f, 0.f, 0.f, 0.f};
        #pragma unroll
        for (int kk = 0; kk < 2; ++kk)
            acc = wmma16(&A2fc[((mt * 2 + kk) * 32 + lane) * 16],
                         &B2  [(kk * 32 + lane) * 16], acc);
        int nl = lane & 15, mh = (lane >> 4) << 3;
        #pragma unroll
        for (int r = 0; r < 8; ++r) {
            int m = mt * 16 + mh + r;
            float v = (nl < 10) ? fmaxf(acc[r] + fb2s[nl], 0.f) : 0.f;
            f3[m][nl] = v;
        }
    }
    __syncthreads();

    // ---------- phase 7: fc3 [10 -> 2], write concat matrix [B, 860] ----------
    if (tid < BT) {
        float o0 = fb3s[0], o1 = fb3s[1];
        #pragma unroll
        for (int j = 0; j < 10; ++j) {
            float v = f3[tid][j];
            o0 = fmaf(v, fw3s[0][j], o0);
            o1 = fmaf(v, fw3s[1][j], o1);
        }
        float* dst = gcat + (size_t)(b0 + tid) * (2 * S_SEQ) + 2 * s;
        dst[0] = o0; dst[1] = o1;
    }
}

// ================= Kernel 2: head MLP 860->256->8->2 with BN =================
__global__ __launch_bounds__(256, 1)
void head_kernel(const float* __restrict__ gcat,
                 const float* __restrict__ hw1, const float* __restrict__ hb1,
                 const float* __restrict__ g1, const float* __restrict__ be1,
                 const float* __restrict__ mu1, const float* __restrict__ va1,
                 const float* __restrict__ hw2, const float* __restrict__ hb2,
                 const float* __restrict__ g2, const float* __restrict__ be2,
                 const float* __restrict__ mu2, const float* __restrict__ va2,
                 const float* __restrict__ hw3, const float* __restrict__ hb3,
                 float* __restrict__ out)
{
    __shared__ __align__(32) _Float16 Abuf[4 * KS2 * 32 * 16];   // cat [64 x 864]
    __shared__ __align__(32) _Float16 Bbuf[2 * KS2 * 32 * 16];   // 32 rows of hw1
    __shared__ __align__(32) _Float16 A2buf[4 * 8 * 32 * 16];    // layer1 out [64 x 256]
    __shared__ __align__(32) _Float16 B2buf[8 * 32 * 16];        // hw2 [16 x 256]
    __shared__ float sc1[256], sh1[256], sc2[8], sh2[8];
    __shared__ float o2[64][16];
    __shared__ float hw3s[16], hb3s[2];

    const int tid  = threadIdx.x;
    const int lane = tid & 31;
    const int wave = tid >> 5;
    const int m0   = blockIdx.x * 64;

    // fold linear bias into BN: y = x*sc + ((hb - mu)*sc + beta)
    { float sc = g1[tid] * rsqrtf(va1[tid] + 1e-5f);
      sc1[tid] = sc; sh1[tid] = (hb1[tid] - mu1[tid]) * sc + be1[tid]; }
    if (tid < 8) { float sc = g2[tid] * rsqrtf(va2[tid] + 1e-5f);
      sc2[tid] = sc; sh2[tid] = (hb2[tid] - mu2[tid]) * sc + be2[tid]; }
    if (tid < 16) hw3s[tid] = hw3[tid];
    if (tid < 2)  hb3s[tid] = hb3[tid];

    for (int m = 0; m < 64; ++m) {
        const float* src = gcat + (size_t)(m0 + m) * 860;
        for (int k = tid; k < 864; k += 256)
            Abuf[a_swz(m, k, KS2)] = (_Float16)((k < 860) ? src[k] : 0.f);
    }
    for (int n = 0; n < 16; ++n) {
        float v = (n < 8) ? hw2[(size_t)n * 256 + tid] : 0.f;
        B2buf[b_swz(n, tid, 8)] = (_Float16)v;
    }
    __syncthreads();

    // layer 1: N=256 in 8 passes of 32 columns; 8 tiles per pass over 8 waves
    for (int ntp = 0; ntp < 8; ++ntp) {
        for (int n2 = 0; n2 < 32; ++n2) {
            const float* wr = hw1 + (size_t)(ntp * 32 + n2) * 860;
            for (int k = tid; k < 864; k += 256)
                Bbuf[b_swz(n2, k, KS2)] = (_Float16)((k < 860) ? wr[k] : 0.f);
        }
        __syncthreads();
        {
            int mt = wave & 3, ntl = wave >> 2;
            v8f acc = {0.f, 0.f, 0.f, 0.f, 0.f, 0.f, 0.f, 0.f};
            for (int kk = 0; kk < KS2; ++kk)
                acc = wmma16(&Abuf[((mt * KS2 + kk) * 32 + lane) * 16],
                             &Bbuf[((ntl * KS2 + kk) * 32 + lane) * 16], acc);
            int nl = lane & 15, mh = (lane >> 4) << 3;
            int n = (ntp * 2 + ntl) * 16 + nl;
            #pragma unroll
            for (int r = 0; r < 8; ++r) {
                int m = mt * 16 + mh + r;
                float v = fmaxf(acc[r] * sc1[n] + sh1[n], 0.f);
                A2buf[a_swz(m, n, 8)] = (_Float16)v;
            }
        }
        __syncthreads();
    }

    // layer 2: [64 x 256] x [256 x 16(pad of 8)]
    if (wave < 4) {
        int mt = wave;
        v8f acc = {0.f, 0.f, 0.f, 0.f, 0.f, 0.f, 0.f, 0.f};
        #pragma unroll
        for (int kk = 0; kk < 8; ++kk)
            acc = wmma16(&A2buf[((mt * 8 + kk) * 32 + lane) * 16],
                         &B2buf[(kk * 32 + lane) * 16], acc);
        int nl = lane & 15, mh = (lane >> 4) << 3;
        #pragma unroll
        for (int r = 0; r < 8; ++r) {
            int m = mt * 16 + mh + r;
            float v = (nl < 8) ? fmaxf(acc[r] * sc2[nl] + sh2[nl], 0.f) : 0.f;
            o2[m][nl] = v;
        }
    }
    __syncthreads();

    if (tid < 64) {
        float a0 = hb3s[0], a1 = hb3s[1];
        #pragma unroll
        for (int j = 0; j < 8; ++j) {
            float v = o2[tid][j];
            a0 = fmaf(v, hw3s[j], a0);
            a1 = fmaf(v, hw3s[8 + j], a1);
        }
        out[(m0 + tid) * 2]     = a0;
        out[(m0 + tid) * 2 + 1] = a1;
    }
}

extern "C" void kernel_launch(void* const* d_in, const int* in_sizes, int n_in,
                              void* d_out, int out_size, void* d_ws, size_t ws_size,
                              hipStream_t stream) {
    const float* X     = (const float*)d_in[0];
    const float* cw1   = (const float*)d_in[1];
    const float* cb1   = (const float*)d_in[2];
    const float* cw2   = (const float*)d_in[3];
    const float* cb2   = (const float*)d_in[4];
    const float* fw1   = (const float*)d_in[5];
    const float* fb1   = (const float*)d_in[6];
    const float* fw2   = (const float*)d_in[7];
    const float* fb2   = (const float*)d_in[8];
    const float* fw3   = (const float*)d_in[9];
    const float* fb3   = (const float*)d_in[10];
    const float* hw1   = (const float*)d_in[11];
    const float* hb1   = (const float*)d_in[12];
    const float* bn1g  = (const float*)d_in[13];
    const float* bn1b  = (const float*)d_in[14];
    const float* bn1m  = (const float*)d_in[15];
    const float* bn1v  = (const float*)d_in[16];
    const float* hw2   = (const float*)d_in[17];
    const float* hb2   = (const float*)d_in[18];
    const float* bn2g  = (const float*)d_in[19];
    const float* bn2b  = (const float*)d_in[20];
    const float* bn2m  = (const float*)d_in[21];
    const float* bn2v  = (const float*)d_in[22];
    const float* hw3   = (const float*)d_in[23];
    const float* hb3   = (const float*)d_in[24];

    float* gcat = (float*)d_ws;  // [512, 860] f32 = 1.76 MB

    dim3 grid1(S_SEQ, 512 / BT);  // 430 x 16 blocks
    seq_net_kernel<<<grid1, 256, 0, stream>>>(X, cw1, cb1, cw2, cb2,
                                              fw1, fb1, fw2, fb2, fw3, fb3, gcat);

    head_kernel<<<512 / 64, 256, 0, stream>>>(gcat, hw1, hb1, bn1g, bn1b, bn1m, bn1v,
                                              hw2, hb2, bn2g, bn2b, bn2m, bn2v,
                                              hw3, hb3, (float*)d_out);
}
